// NNConvSingleArchtiecture_42021960024097
// MI455X (gfx1250) — compile-verified, matrix operand order checked by reading, not run on hardware
//
#include <hip/hip_runtime.h>
#include <hip/hip_bf16.h>

// CDNA5 / gfx1250 NNConv: per-edge MLP fused with einsum as f16-in/f32-acc
// v_wmma_f32_16x16x32_f16 chains. B' weight matrices are pre-swizzled into
// WMMA fragment order (vector ds loads); A fragments built in registers
// (broadcast x * hidden slice, v_pk_mul_f16 co-executing with WMMA).
// Dual accumulators break the WMMA C-chain for deeper pipelining.

typedef __attribute__((ext_vector_type(16))) _Float16 v16h;
typedef __attribute__((ext_vector_type(8)))  _Float16 v8h;
typedef __attribute__((ext_vector_type(8)))  float    v8f;

__device__ __forceinline__ void wave_sched_barrier() {
#if __has_builtin(__builtin_amdgcn_wave_barrier)
    __builtin_amdgcn_wave_barrier();
#endif
}

// k-element index within a fragment for lane-half `khalf`, element j
__host__ __device__ __forceinline__ int frag_k(int khalf, int j) {
    return (j < 8) ? (khalf * 8 + j) : (16 + khalf * 8 + (j - 8));
}

// ---------------------------------------------------------------------------
// Prep: build f16 weights in WMMA *fragment order*.
//  gW1f : [32 lanes][16] — B-fragment of padded [32x16] first-layer weight
//  gBf1 : [9 chunks][2 nt][32 lanes][16] — layer-1 fused B'
//  gBf2 : [17 chunks][1 nt][32 lanes][16] — layer-2 fused B'
// ---------------------------------------------------------------------------
__global__ void prep_kernel(const float* __restrict__ nn1_w1,
                            const float* __restrict__ nn1_w2,
                            const float* __restrict__ nn1_b2,
                            const float* __restrict__ nn2_w1,
                            const float* __restrict__ nn2_w2,
                            const float* __restrict__ nn2_b2,
                            _Float16* __restrict__ gW1f_1,
                            _Float16* __restrict__ gW1f_2,
                            _Float16* __restrict__ gBf1,
                            _Float16* __restrict__ gBf2) {
    int t = blockIdx.x * blockDim.x + threadIdx.x;

    if (t < 512) {  // W1 fragments (both layers)
        int j = t & 15, lane = t >> 4;
        int khalf = lane >> 4, n = lane & 15;
        int kl = frag_k(khalf, j);
        gW1f_1[t] = (_Float16)((kl < 8) ? nn1_w1[kl * 16 + n] : 0.0f);
        gW1f_2[t] = (_Float16)((kl < 8) ? nn2_w1[kl * 16 + n] : 0.0f);
    }
    if (t < 9 * 2 * 32 * 16) {  // layer-1 B' fragments
        int j = t & 15, lane = (t >> 4) & 31, rest = t >> 9;
        int nt = rest & 1, kk = rest >> 1;
        int khalf = lane >> 4, n = lane & 15;
        int K = kk * 32 + frag_k(khalf, j);
        int o = nt * 16 + n;
        float v = 0.0f;
        if (K < 256)      { int i = K >> 4, h = K & 15; v = nn1_w2[h * 512 + i * 32 + o]; }
        else if (K < 272) { int i = K - 256;            v = nn1_b2[i * 32 + o]; }
        gBf1[t] = (_Float16)v;
    }
    if (t < 17 * 1 * 32 * 16) {  // layer-2 B' fragments
        int j = t & 15, lane = (t >> 4) & 31, kk = t >> 9;
        int khalf = lane >> 4, n = lane & 15;
        int K = kk * 32 + frag_k(khalf, j);
        float v;
        if (K < 512) { int i = K >> 4, h = K & 15; v = nn2_w2[h * 512 + i * 16 + n]; }
        else         { int i = K - 512;            v = nn2_b2[i * 16 + n]; }
        gBf2[t] = (_Float16)v;
    }
}

// ---------------------------------------------------------------------------
// Fused edge kernel. One 16-edge tile per wave32:
//   hid = relu(ea @ W1 + b1)                                 (1 WMMA)
//   msg = [x_src(i)*hid(h)]_{K=i*16+h} @ B' (+ bias rows)    (NCHUNK*NT WMMA)
//   atomicAdd msg into agg[dst]
// ---------------------------------------------------------------------------
template <int F_IN, int F_OUT>
__global__ __launch_bounds__(256) void edge_conv_kernel(
        const float*    __restrict__ xin,     // [N, F_IN]
        const float*    __restrict__ eattr,   // [E, 8]
        const int*      __restrict__ srcIdx,  // [E]
        const int*      __restrict__ dstIdx,  // [E]
        const _Float16* __restrict__ gW1f,    // [32][16] fragment order
        const float*    __restrict__ b1,      // [16]
        const _Float16* __restrict__ gBf,     // fragment order B'
        float*          __restrict__ agg,     // [N, F_OUT]
        int E, int nTiles) {
    constexpr int NT     = F_OUT / 16;
    constexpr int NMAIN  = F_IN / 2;          // outer-product chunks
    constexpr int NCHUNK = NMAIN + 1;         // + bias chunk
    constexpr int NW     = 8;
    constexpr int BFRAG  = NCHUNK * NT * 32 * 16;  // halves

    __shared__ alignas(32) _Float16 shB[BFRAG];
    __shared__ alignas(32) _Float16 shH[NW][16][16];

    const int tid = threadIdx.x;
    for (int i = tid; i < BFRAG / 8; i += 256)
        ((uint4*)shB)[i] = ((const uint4*)gBf)[i];
    __syncthreads();

    const int lane = tid & 31;
    const int wv   = tid >> 5;
    const int tile = blockIdx.x * NW + wv;
    if (tile >= nTiles) return;
    const int e0 = tile << 4;

    const int m     = lane & 15;   // A row (edge within tile) / D column
    const int khalf = lane >> 4;   // which K half this lane holds

    // ---- per-lane gathers straight from global ----
    const int eg = min(e0 + m, E - 1);
    const int s  = srcIdx[eg];

    _Float16 xr[F_IN];             // x[src] row in f16 registers
#pragma unroll
    for (int i = 0; i < F_IN; ++i) xr[i] = (_Float16)xin[(size_t)s * F_IN + i];

    float ear[8];                  // edge_attr row
#pragma unroll
    for (int k = 0; k < 8; ++k) ear[k] = eattr[(size_t)eg * 8 + k];

    // ---- hidden = relu(ea @ W1 + b1): one WMMA (K padded 8->32) ----
    v16h aF;
#pragma unroll
    for (int j = 0; j < 8; ++j) {
        aF[j]     = (khalf == 0) ? (_Float16)ear[j] : (_Float16)0.0f;
        aF[j + 8] = (_Float16)0.0f;
    }
    const v16h bF = *(const v16h*)(gW1f + lane * 16);
    v8f ch = {};
    ch = __builtin_amdgcn_wmma_f32_16x16x32_f16(false, aF, false, bF,
                                                (short)0, ch, false, false);
    const float bb = b1[m];        // D layout: this lane holds hidden unit m
#pragma unroll
    for (int r = 0; r < 8; ++r)
        shH[wv][r + 8 * khalf][m] = (_Float16)fmaxf(ch[r] + bb, 0.0f);
    wave_sched_barrier();

    // this lane's hidden slice: hid[m][khalf*8 .. khalf*8+7] (one b128 load)
    const v8h hv = *(const v8h*)&shH[wv][m][khalf * 8];

    // ---- fused msg GEMM: two independent accumulator chains per nt ----
    v8f accA[NT], accB[NT];
#pragma unroll
    for (int nt = 0; nt < NT; ++nt) { accA[nt] = (v8f){}; accB[nt] = (v8f){}; }

#pragma unroll
    for (int kk = 0; kk < NMAIN; ++kk) {
        const _Float16 x0 = xr[2 * kk], x1 = xr[2 * kk + 1];
        v16h a;
#pragma unroll
        for (int j = 0; j < 8; ++j) {
            a[j]     = x0 * hv[j];
            a[j + 8] = x1 * hv[j];
        }
#pragma unroll
        for (int nt = 0; nt < NT; ++nt) {
            const v16h b = *(const v16h*)&shB[((kk * NT + nt) * 32 + lane) * 16];
            if (kk & 1)
                accB[nt] = __builtin_amdgcn_wmma_f32_16x16x32_f16(
                               false, a, false, b, (short)0, accB[nt], false, false);
            else
                accA[nt] = __builtin_amdgcn_wmma_f32_16x16x32_f16(
                               false, a, false, b, (short)0, accA[nt], false, false);
        }
    }
    {   // bias chunk: A rows carry x against b2 rows of B'
        v16h a;
#pragma unroll
        for (int j = 0; j < 16; ++j) {
            int kl = frag_k(khalf, j);
            a[j] = (kl < F_IN) ? xr[kl] : (_Float16)0.0f;
        }
#pragma unroll
        for (int nt = 0; nt < NT; ++nt) {
            const v16h b = *(const v16h*)&shB[((NMAIN * NT + nt) * 32 + lane) * 16];
            accB[nt] = __builtin_amdgcn_wmma_f32_16x16x32_f16(
                           false, a, false, b, (short)0, accB[nt], false, false);
        }
    }

    // ---- combine chains + scatter (D: edge = r + 8*khalf, col = nt*16+m) ----
    int de[8];
#pragma unroll
    for (int r = 0; r < 8; ++r) de[r] = dstIdx[min(e0 + 8 * khalf + r, E - 1)];
#pragma unroll
    for (int nt = 0; nt < NT; ++nt) {
#pragma unroll
        for (int r = 0; r < 8; ++r) {
            if (e0 + 8 * khalf + r < E) {
                float v = accA[nt][r] + accB[nt][r];
                atomicAdd(&agg[(size_t)de[r] * F_OUT + nt * 16 + m], v);
            }
        }
    }
}

// ---------------------------------------------------------------------------
// Node update 1: h1 = relu(x @ root1 + agg1 + bias1)     [N,16]->[N,32]
// (memory-bound: ~32MB traffic; VALU inner product is fine)
// ---------------------------------------------------------------------------
__global__ __launch_bounds__(256) void node1_kernel(
        const float* __restrict__ x, const float* __restrict__ agg1,
        const float* __restrict__ root1, const float* __restrict__ bias1,
        float* __restrict__ h1, int N) {
    __shared__ float sR[16 * 32];
    __shared__ float sB[32];
    for (int i = threadIdx.x; i < 512; i += blockDim.x) sR[i] = root1[i];
    if (threadIdx.x < 32) sB[threadIdx.x] = bias1[threadIdx.x];
    __syncthreads();
    int n = blockIdx.x * blockDim.x + threadIdx.x;
    if (n >= N) return;
    float xr[16];
#pragma unroll
    for (int i = 0; i < 16; ++i) xr[i] = x[(size_t)n * 16 + i];
    for (int o = 0; o < 32; ++o) {
        float t = sB[o] + agg1[(size_t)n * 32 + o];
#pragma unroll
        for (int i = 0; i < 16; ++i) t = fmaf(xr[i], sR[i * 32 + o], t);
        h1[(size_t)n * 32 + o] = fmaxf(t, 0.0f);
    }
}

// ---------------------------------------------------------------------------
// Node update 2 + graph pooling accumulation.
// ---------------------------------------------------------------------------
__global__ __launch_bounds__(256) void node2_kernel(
        const float* __restrict__ h1, const float* __restrict__ agg2,
        const float* __restrict__ root2, const float* __restrict__ bias2,
        const int* __restrict__ batch,
        float* __restrict__ gsum, float* __restrict__ gcnt, int N) {
    __shared__ float sR[32 * 16];
    __shared__ float sB[16];
    for (int i = threadIdx.x; i < 512; i += blockDim.x) sR[i] = root2[i];
    if (threadIdx.x < 16) sB[threadIdx.x] = bias2[threadIdx.x];
    __syncthreads();
    int n = blockIdx.x * blockDim.x + threadIdx.x;
    if (n >= N) return;
    float hr[32];
#pragma unroll
    for (int i = 0; i < 32; ++i) hr[i] = h1[(size_t)n * 32 + i];
    int g = batch[n];
    atomicAdd(&gcnt[g], 1.0f);
    for (int o = 0; o < 16; ++o) {
        float t = sB[o] + agg2[(size_t)n * 16 + o];
#pragma unroll
        for (int i = 0; i < 32; ++i) t = fmaf(hr[i], sR[i * 16 + o], t);
        atomicAdd(&gsum[(size_t)g * 16 + o], fmaxf(t, 0.0f));
    }
}

// ---------------------------------------------------------------------------
// Readout: pooled = sums/max(cnt,1); out = relu(pooled@l1+b1)@l2+b2
// ---------------------------------------------------------------------------
__global__ __launch_bounds__(256) void readout_kernel(
        const float* __restrict__ gsum, const float* __restrict__ gcnt,
        const float* __restrict__ l1w, const float* __restrict__ l1b,
        const float* __restrict__ l2w, const float* __restrict__ l2b,
        float* __restrict__ out, int G) {
    int g = blockIdx.x * blockDim.x + threadIdx.x;
    if (g >= G) return;
    float inv = 1.0f / fmaxf(gcnt[g], 1.0f);
    float p[16];
#pragma unroll
    for (int o = 0; o < 16; ++o) p[o] = gsum[(size_t)g * 16 + o] * inv;
    float acc = l2b[0];
    for (int j = 0; j < 8; ++j) {
        float t = l1b[j];
#pragma unroll
        for (int o = 0; o < 16; ++o) t = fmaf(p[o], l1w[o * 8 + j], t);
        acc = fmaf(fmaxf(t, 0.0f), l2w[j], acc);
    }
    out[g] = acc;
}

// ---------------------------------------------------------------------------
extern "C" void kernel_launch(void* const* d_in, const int* in_sizes, int n_in,
                              void* d_out, int out_size, void* d_ws, size_t ws_size,
                              hipStream_t stream) {
    const float* x       = (const float*)d_in[0];
    const float* eattr   = (const float*)d_in[1];
    const int*   eidx    = (const int*)  d_in[2];
    const int*   batch   = (const int*)  d_in[3];
    const float* nn1_w1  = (const float*)d_in[4];
    const float* nn1_b1  = (const float*)d_in[5];
    const float* nn1_w2  = (const float*)d_in[6];
    const float* nn1_b2  = (const float*)d_in[7];
    const float* root1   = (const float*)d_in[8];
    const float* bias1   = (const float*)d_in[9];
    const float* nn2_w1  = (const float*)d_in[10];
    const float* nn2_b1  = (const float*)d_in[11];
    const float* nn2_w2  = (const float*)d_in[12];
    const float* nn2_b2  = (const float*)d_in[13];
    const float* root2   = (const float*)d_in[14];
    const float* bias2   = (const float*)d_in[15];
    const float* lin1_w  = (const float*)d_in[16];
    const float* lin1_b  = (const float*)d_in[17];
    const float* lin2_w  = (const float*)d_in[18];
    const float* lin2_b  = (const float*)d_in[19];
    float* out = (float*)d_out;

    const int N = in_sizes[0] / 16;   // nodes
    const int E = in_sizes[2] / 2;    // edges
    const int G = out_size;           // graphs

    char* base = (char*)d_ws;
    size_t off = 0;
    auto alloc = [&](size_t bytes) {
        char* p = base + off;
        off = (off + bytes + 255) & ~(size_t)255;
        return p;
    };
    float*    agg1   = (float*)   alloc((size_t)N * 32 * 4);
    float*    h1     = (float*)   alloc((size_t)N * 32 * 4);
    float*    agg2   = (float*)   alloc((size_t)N * 16 * 4);
    float*    gsum   = (float*)   alloc((size_t)G * 16 * 4);
    float*    gcnt   = (float*)   alloc((size_t)G * 4);
    _Float16* gW1f_1 = (_Float16*)alloc(512 * 2);
    _Float16* gW1f_2 = (_Float16*)alloc(512 * 2);
    _Float16* gBf1   = (_Float16*)alloc((size_t)9 * 2 * 32 * 16 * 2);
    _Float16* gBf2   = (_Float16*)alloc((size_t)17 * 1 * 32 * 16 * 2);
    (void)ws_size; (void)n_in;

    hipMemsetAsync(agg1, 0, (size_t)N * 32 * 4, stream);
    hipMemsetAsync(agg2, 0, (size_t)N * 16 * 4, stream);
    hipMemsetAsync(gsum, 0, (size_t)G * 16 * 4, stream);
    hipMemsetAsync(gcnt, 0, (size_t)G * 4, stream);

    prep_kernel<<<(9 * 2 * 32 * 16 + 255) / 256, 256, 0, stream>>>(
        nn1_w1, nn1_w2, nn1_b2, nn2_w1, nn2_w2, nn2_b2,
        gW1f_1, gW1f_2, gBf1, gBf2);

    const int nTiles  = (E + 15) / 16;
    const int eBlocks = (nTiles + 7) / 8;
    const int* srcIdx = eidx;
    const int* dstIdx = eidx + E;

    edge_conv_kernel<16, 32><<<eBlocks, 256, 0, stream>>>(
        x, eattr, srcIdx, dstIdx, gW1f_1, nn1_b1, gBf1, agg1, E, nTiles);

    node1_kernel<<<(N + 255) / 256, 256, 0, stream>>>(
        x, agg1, root1, bias1, h1, N);

    edge_conv_kernel<32, 16><<<eBlocks, 256, 0, stream>>>(
        h1, eattr, srcIdx, dstIdx, gW1f_2, nn2_b1, gBf2, agg2, E, nTiles);

    node2_kernel<<<(N + 255) / 256, 256, 0, stream>>>(
        h1, agg2, root2, bias2, batch, gsum, gcnt, N);

    readout_kernel<<<(G + 255) / 256, 256, 0, stream>>>(
        gsum, gcnt, lin1_w, lin1_b, lin2_w, lin2_b, out, G);
}
